// DeeperGCN_65317862637686
// MI455X (gfx1250) — matrix operand order repaired and checked by reading, not compiled, not run on hardware
//
#include <hip/hip_runtime.h>
#include <hip/hip_bf16.h>
#include <float.h>

// Problem constants (match reference)
#define NB   32          // samples
#define NPS  1024        // nodes per sample
#define NN   32768       // total nodes (B*N)
#define H    64          // hidden
#define H2   128         // 2*H
#define EPS_ ( (int)16384 )   // edges per sample (power of 2)
#define BE   524288      // total edges
#define NL   4           // layers
#define LNE  1e-5f

typedef __attribute__((ext_vector_type(16))) __bf16 v16bf;
typedef __attribute__((ext_vector_type(8)))  float  v8f;

// ---------------------------------------------------------------- helpers
__device__ __forceinline__ void atomicMaxF(float* addr, float val) {
  if (val >= 0.0f) atomicMax((int*)addr, __float_as_int(val));
  else             atomicMin((unsigned int*)addr, __float_as_uint(val));
}

__device__ __forceinline__ float waveAllSum(float v) {
#pragma unroll
  for (int m = 16; m >= 1; m >>= 1) v += __shfl_xor(v, m, 32);
  return v;
}

// ---------------------------------------------------------------- encoders
__global__ void enc_nodes(const float* __restrict__ x, const float* __restrict__ w,
                          const float* __restrict__ b, float* __restrict__ h) {
  int idx = blockIdx.x * blockDim.x + threadIdx.x;  // NN*H
  int n = idx >> 6, f = idx & 63;
  h[idx] = x[n] * w[f] + b[f];
}

__global__ void enc_edges(const float* __restrict__ ea_in, const float* __restrict__ w,
                          const float* __restrict__ b, float* __restrict__ ea) {
  int idx = blockIdx.x * blockDim.x + threadIdx.x;  // EPS_*H
  int e = idx >> 6, f = idx & 63;
  ea[idx] = ea_in[e] * w[f] + b[f];
}

// ---------------------------------------------------------------- LN kernels
// z = relu(LN(h; g,b)) over H=64 features, one wave per node (2 feats/lane)
__global__ void ln_relu64(const float* __restrict__ Hs, const float* __restrict__ g,
                          const float* __restrict__ bb, float* __restrict__ Z) {
  int wid = (blockIdx.x * blockDim.x + threadIdx.x) >> 5;
  int lane = threadIdx.x & 31;
  const float* row = Hs + (size_t)wid * H;
  float x0 = row[lane], x1 = row[lane + 32];
  float mu = waveAllSum(x0 + x1) * (1.0f / 64.0f);
  float d0 = x0 - mu, d1 = x1 - mu;
  float var = waveAllSum(d0 * d0 + d1 * d1) * (1.0f / 64.0f);
  float r = rsqrtf(var + LNE);
  Z[(size_t)wid * H + lane]      = fmaxf(d0 * r * g[lane]      + bb[lane],      0.0f);
  Z[(size_t)wid * H + lane + 32] = fmaxf(d1 * r * g[lane + 32] + bb[lane + 32], 0.0f);
}

// in-place u = relu(LN(u; g,b)) over 2H=128 features, one wave per node
__global__ void ln_relu128(float* __restrict__ U, const float* __restrict__ g,
                           const float* __restrict__ bb) {
  int wid = (blockIdx.x * blockDim.x + threadIdx.x) >> 5;
  int lane = threadIdx.x & 31;
  float* row = U + (size_t)wid * H2;
  float x[4], s = 0.0f;
#pragma unroll
  for (int j = 0; j < 4; ++j) { x[j] = row[lane + 32 * j]; s += x[j]; }
  float mu = waveAllSum(s) * (1.0f / 128.0f);
  float vs = 0.0f;
#pragma unroll
  for (int j = 0; j < 4; ++j) { float d = x[j] - mu; vs += d * d; }
  float r = rsqrtf(waveAllSum(vs) * (1.0f / 128.0f) + LNE);
#pragma unroll
  for (int j = 0; j < 4; ++j) {
    int f = lane + 32 * j;
    row[f] = fmaxf((x[j] - mu) * r * g[f] + bb[f], 0.0f);
  }
}

// ---------------------------------------------------------------- misc elementwise
__global__ void copy_f(const float* __restrict__ a, float* __restrict__ d) {
  int i = blockIdx.x * blockDim.x + threadIdx.x; d[i] = a[i];
}
__global__ void init_mda(float* __restrict__ m, float* __restrict__ den, float* __restrict__ agg) {
  int i = blockIdx.x * blockDim.x + threadIdx.x;
  m[i] = -FLT_MAX; den[i] = 0.0f; agg[i] = 0.0f;
}
__global__ void add_z(float* __restrict__ agg, const float* __restrict__ z) {
  int i = blockIdx.x * blockDim.x + threadIdx.x; agg[i] += z[i];
}
__global__ void hid_init(float* __restrict__ hid, const float* __restrict__ b) {
  int i = blockIdx.x * blockDim.x + threadIdx.x;
  if (i < NB * 1000) hid[i] = b[i % 1000];
}

// ---------------------------------------------------------------- edge softmax-aggr passes
// thread per (edge, feature); msg recomputed each pass (bandwidth-bound by design)
__global__ void edge_max(const int* __restrict__ src, const int* __restrict__ dst,
                         const float* __restrict__ z, const float* __restrict__ ea,
                         const float* __restrict__ t, int layer, float* __restrict__ m) {
  int idx = blockIdx.x * blockDim.x + threadIdx.x;          // BE*H
  int e = idx >> 6, f = idx & 63;
  float msg = fmaxf(z[(size_t)src[e] * H + f] + ea[(size_t)(e & (EPS_ - 1)) * H + f], 0.0f) + 1e-7f;
  atomicMaxF(&m[(size_t)dst[e] * H + f], msg * t[layer]);
}
__global__ void edge_den(const int* __restrict__ src, const int* __restrict__ dst,
                         const float* __restrict__ z, const float* __restrict__ ea,
                         const float* __restrict__ t, int layer,
                         const float* __restrict__ m, float* __restrict__ den) {
  int idx = blockIdx.x * blockDim.x + threadIdx.x;
  int e = idx >> 6, f = idx & 63;
  int d = dst[e];
  float msg = fmaxf(z[(size_t)src[e] * H + f] + ea[(size_t)(e & (EPS_ - 1)) * H + f], 0.0f) + 1e-7f;
  atomicAdd(&den[(size_t)d * H + f], __expf(msg * t[layer] - m[(size_t)d * H + f]));
}
__global__ void edge_agg(const int* __restrict__ src, const int* __restrict__ dst,
                         const float* __restrict__ z, const float* __restrict__ ea,
                         const float* __restrict__ t, int layer,
                         const float* __restrict__ m, const float* __restrict__ den,
                         float* __restrict__ agg) {
  int idx = blockIdx.x * blockDim.x + threadIdx.x;
  int e = idx >> 6, f = idx & 63;
  int d = dst[e];
  float msg = fmaxf(z[(size_t)src[e] * H + f] + ea[(size_t)(e & (EPS_ - 1)) * H + f], 0.0f) + 1e-7f;
  float lg = msg * t[layer];
  float alpha = __expf(lg - m[(size_t)d * H + f]) / (den[(size_t)d * H + f] + 1e-16f);
  atomicAdd(&agg[(size_t)d * H + f], msg * alpha);
}

// ---------------------------------------------------------------- WMMA GEMMs (bf16 -> f32)
// Operand packing per CDNA5 ISA:
//  A 16x32 bf16: lane L (M=L&15, hi=L>>4): vgpr v<4 -> K=2v+8*hi{,+1}; v>=4 -> K=16+2(v-4)+8*hi{,+1}
//  B 32x16 bf16: lane L (N=L&15, hi=L>>4): vgpr v<4 -> K=2v+8*hi{,+1}; v>=4 -> K=16+2(v-4)+8*hi{,+1}
//  C/D 16x16 f32: vgpr r, lane L -> M=r+8*(L>>4), N=L&15

// upre[NN,128] = out[NN,64] @ w1[128,64]^T + b1   (one 16x16 tile per wave, K=64)
__global__ void wmma_gemm1(const float* __restrict__ A, const float* __restrict__ W,
                           const float* __restrict__ bias, float* __restrict__ D) {
  int wid = (blockIdx.x * blockDim.x + threadIdx.x) >> 5;
  int lane = threadIdx.x & 31;
  int mt = wid >> 3, nt = wid & 7;               // 2048 x 8 tiles
  int m0 = mt << 4, n0 = nt << 4;
  int hi = lane >> 4, lm = lane & 15;
  float bv = bias[n0 + lm];
  v8f c;
#pragma unroll
  for (int r = 0; r < 8; ++r) c[r] = bv;
  const float* Arow = A + (size_t)(m0 + lm) * H;
  const float* Wrow = W + (size_t)(n0 + lm) * H;  // w1[j,:], j = n0+lm
#pragma unroll
  for (int ks = 0; ks < H; ks += 32) {
    v16bf a, b;
#pragma unroll
    for (int v = 0; v < 8; ++v) {
      int ka = ((v < 4) ? (2 * v) : (16 + 2 * (v - 4))) + 8 * hi + ks;
      a[2 * v]     = (__bf16)Arow[ka];
      a[2 * v + 1] = (__bf16)Arow[ka + 1];
      int kb = ((v < 4) ? (2 * v) : (16 + 2 * (v - 4))) + 8 * hi + ks;
      b[2 * v]     = (__bf16)Wrow[kb];
      b[2 * v + 1] = (__bf16)Wrow[kb + 1];
    }
    c = __builtin_amdgcn_wmma_f32_16x16x32_bf16(false, a, false, b, (short)0, c, false, false);
  }
#pragma unroll
  for (int r = 0; r < 8; ++r)
    D[(size_t)(m0 + r + 8 * hi) * H2 + n0 + lm] = c[r];
}

// h[NN,64] = (residual ? h : 0) + u[NN,128] @ w2[64,128]^T + b2   (K=128)
__global__ void wmma_gemm2(const float* __restrict__ U, const float* __restrict__ W,
                           const float* __restrict__ bias, float* __restrict__ Ho, int residual) {
  int wid = (blockIdx.x * blockDim.x + threadIdx.x) >> 5;
  int lane = threadIdx.x & 31;
  int mt = wid >> 2, nt = wid & 3;               // 2048 x 4 tiles
  int m0 = mt << 4, n0 = nt << 4;
  int hi = lane >> 4, lm = lane & 15;
  float bv = bias[n0 + lm];
  v8f c;
#pragma unroll
  for (int r = 0; r < 8; ++r) {
    c[r] = bv;
    if (residual) c[r] += Ho[(size_t)(m0 + r + 8 * hi) * H + n0 + lm];
  }
  const float* Arow = U + (size_t)(m0 + lm) * H2;
  const float* Wrow = W + (size_t)(n0 + lm) * H2;
#pragma unroll
  for (int ks = 0; ks < H2; ks += 32) {
    v16bf a, b;
#pragma unroll
    for (int v = 0; v < 8; ++v) {
      int ka = ((v < 4) ? (2 * v) : (16 + 2 * (v - 4))) + 8 * hi + ks;
      a[2 * v]     = (__bf16)Arow[ka];
      a[2 * v + 1] = (__bf16)Arow[ka + 1];
      int kb = ((v < 4) ? (2 * v) : (16 + 2 * (v - 4))) + 8 * hi + ks;
      b[2 * v]     = (__bf16)Wrow[kb];
      b[2 * v + 1] = (__bf16)Wrow[kb + 1];
    }
    c = __builtin_amdgcn_wmma_f32_16x16x32_bf16(false, a, false, b, (short)0, c, false, false);
  }
#pragma unroll
  for (int r = 0; r < 8; ++r)
    Ho[(size_t)(m0 + r + 8 * hi) * H + n0 + lm] = c[r];
}

// hid[32,1008) += hflat[32,65536] @ lin1_w[1000,65536]^T
// K split over 64 chunks of 1024; each wave computes BOTH 16-row M tiles so the
// dominant HBM stream (lin1_w, 262 MB, > L2) is read exactly once. The shared B
// fragment feeds two back-to-back WMMAs.
__global__ void wmma_lin1(const float* __restrict__ A, const float* __restrict__ W,
                          float* __restrict__ hid) {
  int wid = (blockIdx.x * blockDim.x + threadIdx.x) >> 5;  // 63*64 = 4032 waves
  int lane = threadIdx.x & 31;
  int nt = wid % 63;
  int kc = wid / 63;
  int n0 = nt << 4, k0 = kc << 10;
  int hi = lane >> 4, lm = lane & 15;
  int ocol = n0 + lm;
  bool valid = ocol < 1000;
  const float* Arow0 = A + (size_t)lm * 65536;         // rows 0..15
  const float* Arow1 = A + (size_t)(16 + lm) * 65536;  // rows 16..31
  const float* Wrow  = W + (size_t)(valid ? ocol : 0) * 65536;
  v8f c0, c1;
#pragma unroll
  for (int r = 0; r < 8; ++r) { c0[r] = 0.0f; c1[r] = 0.0f; }
  for (int ks = k0; ks < k0 + 1024; ks += 32) {
    // prefetch weight stream one chunk ahead (emits global_prefetch_b8)
    __builtin_prefetch(Wrow + ks + 256, 0, 1);
    v16bf a0, a1, b;
#pragma unroll
    for (int v = 0; v < 8; ++v) {
      int ka = ((v < 4) ? (2 * v) : (16 + 2 * (v - 4))) + 8 * hi + ks;
      a0[2 * v]     = (__bf16)Arow0[ka];
      a0[2 * v + 1] = (__bf16)Arow0[ka + 1];
      a1[2 * v]     = (__bf16)Arow1[ka];
      a1[2 * v + 1] = (__bf16)Arow1[ka + 1];
      int kb = ((v < 4) ? (2 * v) : (16 + 2 * (v - 4))) + 8 * hi + ks;
      float w0 = valid ? Wrow[kb]     : 0.0f;
      float w1 = valid ? Wrow[kb + 1] : 0.0f;
      b[2 * v]     = (__bf16)w0;
      b[2 * v + 1] = (__bf16)w1;
    }
    c0 = __builtin_amdgcn_wmma_f32_16x16x32_bf16(false, a0, false, b, (short)0, c0, false, false);
    c1 = __builtin_amdgcn_wmma_f32_16x16x32_bf16(false, a1, false, b, (short)0, c1, false, false);
  }
#pragma unroll
  for (int r = 0; r < 8; ++r) {
    if (valid) {
      atomicAdd(&hid[(size_t)(r + 8 * hi) * 1000 + ocol],      c0[r]);
      atomicAdd(&hid[(size_t)(16 + r + 8 * hi) * 1000 + ocol], c1[r]);
    }
  }
}

// out[b] = lin2_b + sum_o relu(hid[b,o]) * lin2_w[o]
__global__ void final_out(const float* __restrict__ hid, const float* __restrict__ w,
                          const float* __restrict__ b, float* __restrict__ out) {
  int bi = blockIdx.x;
  float s = 0.0f;
  for (int o = threadIdx.x; o < 1000; o += 256)
    s += fmaxf(hid[(size_t)bi * 1000 + o], 0.0f) * w[o];
  __shared__ float red[256];
  red[threadIdx.x] = s; __syncthreads();
  for (int st = 128; st > 0; st >>= 1) {
    if (threadIdx.x < st) red[threadIdx.x] += red[threadIdx.x + st];
    __syncthreads();
  }
  if (threadIdx.x == 0) out[bi] = red[0] + b[0];
}

// ---------------------------------------------------------------- launch
extern "C" void kernel_launch(void* const* d_in, const int* in_sizes, int n_in,
                              void* d_out, int out_size, void* d_ws, size_t ws_size,
                              hipStream_t stream) {
  const float* x        = (const float*)d_in[0];
  const int*   eidx     = (const int*)  d_in[1];   // [2, BE]
  const float* eattr    = (const float*)d_in[2];
  // d_in[3] = batch (unused)
  const float* node_w   = (const float*)d_in[4];
  const float* node_b   = (const float*)d_in[5];
  const float* edge_w   = (const float*)d_in[6];
  const float* edge_b   = (const float*)d_in[7];
  const float* t        = (const float*)d_in[8];
  const float* w1       = (const float*)d_in[9];   // [L,128,64]
  const float* b1       = (const float*)d_in[10];  // [L,128]
  const float* g1       = (const float*)d_in[11];
  const float* be1      = (const float*)d_in[12];
  const float* w2       = (const float*)d_in[13];  // [L,64,128]
  const float* b2       = (const float*)d_in[14];  // [L,64]
  const float* ln_g     = (const float*)d_in[15];  // [L,64]
  const float* ln_b     = (const float*)d_in[16];
  const float* lin1_w   = (const float*)d_in[17];  // [1000,65536]
  const float* lin1_b   = (const float*)d_in[18];
  const float* lin2_w   = (const float*)d_in[19];
  const float* lin2_b   = (const float*)d_in[20];
  float* out = (float*)d_out;

  const int* src = eidx;
  const int* dst = eidx + BE;

  // workspace carving
  float* h   = (float*)d_ws;            // NN*H
  float* z   = h   + (size_t)NN * H;    // NN*H
  float* ea  = z   + (size_t)NN * H;    // EPS_*H
  float* m   = ea  + (size_t)EPS_ * H;  // NN*H
  float* den = m   + (size_t)NN * H;    // NN*H
  float* agg = den + (size_t)NN * H;    // NN*H
  float* u   = agg + (size_t)NN * H;    // NN*H2
  float* hid = u   + (size_t)NN * H2;   // 32*1000

  const int TB = 256;
  dim3 blkNH((NN * H) / TB);            // 8192 blocks, elementwise over nodes*feats
  dim3 blkEH((EPS_ * H) / TB);          // 4096
  dim3 blkEdge((BE * 64) / TB);         // 131072, thread per (edge,feat)
  dim3 blkWaveNode((NN * 32) / TB);     // one wave per node -> 4096 blocks

  enc_nodes<<<blkNH, TB, 0, stream>>>(x, node_w, node_b, h);
  enc_edges<<<blkEH, TB, 0, stream>>>(eattr, edge_w, edge_b, ea);

  for (int l = 0; l < NL; ++l) {
    if (l == 0) copy_f<<<blkNH, TB, 0, stream>>>(h, z);
    else        ln_relu64<<<blkWaveNode, TB, 0, stream>>>(h, ln_g + l * H, ln_b + l * H, z);

    init_mda<<<blkNH, TB, 0, stream>>>(m, den, agg);
    edge_max<<<blkEdge, TB, 0, stream>>>(src, dst, z, ea, t, l, m);
    edge_den<<<blkEdge, TB, 0, stream>>>(src, dst, z, ea, t, l, m, den);
    edge_agg<<<blkEdge, TB, 0, stream>>>(src, dst, z, ea, t, l, m, den, agg);
    add_z<<<blkNH, TB, 0, stream>>>(agg, z);

    // MLP: u = relu(LN(agg @ w1^T + b1)); h = (l? h:0) + u @ w2^T + b2
    wmma_gemm1<<<dim3((2048 * 8 * 32) / TB), TB, 0, stream>>>(agg, w1 + (size_t)l * H2 * H,
                                                              b1 + l * H2, u);
    ln_relu128<<<blkWaveNode, TB, 0, stream>>>(u, g1 + l * H2, be1 + l * H2);
    wmma_gemm2<<<dim3((2048 * 4 * 32) / TB), TB, 0, stream>>>(u, w2 + (size_t)l * H * H2,
                                                              b2 + l * H, h, l > 0 ? 1 : 0);
  }

  // final norm (layer-0 params) into z, then readout
  ln_relu64<<<blkWaveNode, TB, 0, stream>>>(h, ln_g, ln_b, z);
  hid_init<<<dim3((NB * 1000 + TB - 1) / TB), TB, 0, stream>>>(hid, lin1_b);
  wmma_lin1<<<dim3((4032 * 32) / TB), TB, 0, stream>>>(z, lin1_w, hid);
  final_out<<<dim3(NB), TB, 0, stream>>>(hid, lin2_w, lin2_b, out);
}
